// WorldModel_16621523435805
// MI455X (gfx1250) — matrix operand order repaired and checked by reading, not compiled, not run on hardware
//
#include <hip/hip_runtime.h>
#include <cstdint>
#include <cstddef>

// ---------------------------------------------------------------------------
// MI455X (gfx1250) world-model scan.
// Strategy: pre-cast all weights to bf16 (~161MB -> resident in 192MB L2 for
// the whole 64-step scan), run every GEMM on v_wmma_f32_16x16x32_bf16 with
// B=16 == one WMMA M tile. A fragments loaded straight from global with
// 2x global_load_b128 per lane (row-major layout == fragment layout),
// B tiles pre-packed in fragment order (1 contiguous 32B read per lane).
// ---------------------------------------------------------------------------

typedef __attribute__((ext_vector_type(16))) __bf16 v16bf;
typedef __attribute__((ext_vector_type(8)))  __bf16 v8bf;
typedef __attribute__((ext_vector_type(8)))  float  v8f;

namespace wm {
constexpr int B_ = 16, T_ = 64, IN_ = 1024, R_ = 4096, A_ = 18;
constexpr int NC_ = 32, NK_ = 32, LAT_ = 1024, NB_ = 8, BS_ = 512;
constexpr int FULL_ = R_ + LAT_;        // 5120
constexpr int GIN_  = FULL_ + A_;       // 5138
constexpr int G3_   = 3 * BS_;          // 1536
constexpr int NTOTG_ = NB_ * G3_;       // 12288
// d_out layout (floats)
constexpr size_t OFF_FULLS = 0;
constexpr size_t OFF_RECS  = (size_t)B_ * T_ * FULL_;               // 5,242,880
constexpr size_t OFF_POST  = OFF_RECS + (size_t)B_ * T_ * R_;       // 9,437,184
constexpr size_t OFF_PRIOR = OFF_POST + (size_t)B_ * T_ * LAT_;     // 10,485,760
}

__device__ __forceinline__ float sigf(float x) { return 1.0f / (1.0f + expf(-x)); }

// ---------------------------------------------------------------------------
// Pack f32 weights into bf16 WMMA B-fragment tile order.
// Tile (kt, nt) holds 512 bf16: idx = lane*16 + j with
//   K = kt*32 + (lane>>4)*8 + (j&7) + (j>>3)*16 ,  N = nt*16 + (lane&15)
// Source element (k, n): blk = n / nPerBlk, col = n % nPerBlk,
//   W[blk*blkStride + k*ldw + col]   (blkStride=0 for plain matrices)
// ---------------------------------------------------------------------------
__global__ void k_pack_b(const float* __restrict__ W, __bf16* __restrict__ out,
                         int K, int Ntot, int nPerBlk, long long blkStride, int ldw)
{
    size_t gid = (size_t)blockIdx.x * blockDim.x + threadIdx.x;
    size_t total = (size_t)K * Ntot;
    if (gid >= total) return;
    int nTiles = Ntot >> 4;
    size_t tile = gid >> 9;
    int idx = (int)(gid & 511);
    int kt = (int)(tile / nTiles);
    int nt = (int)(tile % nTiles);
    int l = idx >> 4, j = idx & 15;
    int k = kt * 32 + ((l >> 4) << 3) + (j & 7) + ((j >> 3) << 4);
    int n = nt * 16 + (l & 15);
    int blk = n / nPerBlk, col = n % nPerBlk;
    out[gid] = (__bf16)W[(size_t)blk * blkStride + (size_t)k * ldw + col];
}

// ---------------------------------------------------------------------------
// Wave-level WMMA GEMM: out[M,Ntot] = act(A[M,K] @ Bpack + bias)
// blockDim=256 (8 independent waves). Wave w owns N-tile (blockIdx.x*8+w),
// row tile blockIdx.y. A-fragment per lane = two contiguous 16B segments of
// the row-major A matrix -> 2x global_load_b128, no LDS/barriers needed.
// K unrolled x2 with split accumulators to break the WMMA RAW chain.
// For the GRU recurrent GEMM, A columns are offset per N-block (aColStride).
// ---------------------------------------------------------------------------
template <bool SILU_BF16>
__global__ void k_gemm_wmma(const __bf16* __restrict__ A, int lda,
                            const __bf16* __restrict__ Bpack,
                            const float* __restrict__ bias,
                            void* __restrict__ outv,
                            int K, int Ntot, int ntPerBlk, int aColStride)
{
    const int lane = threadIdx.x & 31;
    const int wave = threadIdx.x >> 5;
    const int ntGlobal = blockIdx.x * 8 + wave;
    const int mbase = blockIdx.y * 16;
    const int nTiles = Ntot >> 4;
    const int blk = (blockIdx.x * 8) / (ntPerBlk >> 4);   // same for all 8 waves
    const int aOff = blk * aColStride;
    const int kTiles = K >> 5;
    const int hi = lane >> 4;

    // Per-lane A row base: fragment = A[m][k0 + hi*8 .. +8) and [k0+16+hi*8 ..)
    const __bf16* Arow = A + (size_t)(mbase + (lane & 15)) * lda + aOff + (hi << 3);
    const __bf16* Btile = Bpack + ((size_t)ntGlobal << 9) + (lane << 4);
    const size_t bKStride = (size_t)nTiles << 9;

    v8f acc0 = {}, acc1 = {};
    int kt = 0;
    for (; kt + 1 < kTiles; kt += 2) {
        v8bf a0lo = *(const v8bf*)(Arow + kt * 32);
        v8bf a0hi = *(const v8bf*)(Arow + kt * 32 + 16);
        v16bf av0 = __builtin_shufflevector(a0lo, a0hi, 0,1,2,3,4,5,6,7,8,9,10,11,12,13,14,15);
        v16bf bv0 = *(const v16bf*)(Btile + (size_t)kt * bKStride);
        v8bf a1lo = *(const v8bf*)(Arow + (kt + 1) * 32);
        v8bf a1hi = *(const v8bf*)(Arow + (kt + 1) * 32 + 16);
        v16bf av1 = __builtin_shufflevector(a1lo, a1hi, 0,1,2,3,4,5,6,7,8,9,10,11,12,13,14,15);
        v16bf bv1 = *(const v16bf*)(Btile + (size_t)(kt + 1) * bKStride);
        if (kt + 2 < kTiles)
            __builtin_prefetch(Btile + (size_t)(kt + 2) * bKStride, 0, 1);
        acc0 = __builtin_amdgcn_wmma_f32_16x16x32_bf16(false, av0, false, bv0,
                                                       (short)0, acc0, false, false);
        acc1 = __builtin_amdgcn_wmma_f32_16x16x32_bf16(false, av1, false, bv1,
                                                       (short)0, acc1, false, false);
    }
    if (kt < kTiles) {
        v8bf alo = *(const v8bf*)(Arow + kt * 32);
        v8bf ahi = *(const v8bf*)(Arow + kt * 32 + 16);
        v16bf av = __builtin_shufflevector(alo, ahi, 0,1,2,3,4,5,6,7,8,9,10,11,12,13,14,15);
        v16bf bv = *(const v16bf*)(Btile + (size_t)kt * bKStride);
        acc0 = __builtin_amdgcn_wmma_f32_16x16x32_bf16(false, av, false, bv,
                                                       (short)0, acc0, false, false);
    }
    v8f acc = acc0 + acc1;

    const int ncol = ntGlobal * 16 + (lane & 15);
    const float bvs = bias ? bias[ncol] : 0.0f;
#pragma unroll
    for (int v = 0; v < 8; ++v) {
        int row = mbase + hi * 8 + v;
        float val = acc[v] + bvs;
        if (SILU_BF16) {
            float s = val * sigf(val);
            ((__bf16*)outv)[(size_t)row * Ntot + ncol] = (__bf16)s;
        } else {
            ((float*)outv)[(size_t)row * Ntot + ncol] = val;
        }
    }
}

// ---- concat(h, obs_t) -> bf16 [16,5120] --------------------------------
__global__ void k_build_xpost(const float* __restrict__ h,
                              const float* __restrict__ obs, int t,
                              __bf16* __restrict__ xpost)
{
    int gid = blockIdx.x * blockDim.x + threadIdx.x;
    if (gid >= wm::B_ * wm::FULL_) return;
    int b = gid / wm::FULL_, c = gid % wm::FULL_;
    float v = (c < wm::R_) ? h[b * wm::R_ + c]
                           : obs[((size_t)(b * wm::T_ + t)) * wm::IN_ + (c - wm::R_)];
    xpost[gid] = (__bf16)v;
}

// ---- softmax / log-softmax / categorical sample / latent one-hot -------
__global__ void k_sample(const float* __restrict__ logits, int t,
                         float* __restrict__ post_out,   // d_out + OFF_POST
                         float* __restrict__ fulls_out,  // d_out + OFF_FULLS
                         __bf16* __restrict__ xfull)
{
    int tid = blockIdx.x * blockDim.x + threadIdx.x;
    if (tid >= wm::B_ * wm::NC_) return;
    int b = tid / wm::NC_, c = tid % wm::NC_;
    const float* lg = logits + b * wm::LAT_ + c * wm::NK_;
    float mx = lg[0];
    for (int k = 1; k < wm::NK_; ++k) mx = fmaxf(mx, lg[k]);
    float ex[wm::NK_], sum = 0.0f;
    for (int k = 0; k < wm::NK_; ++k) { ex[k] = expf(lg[k] - mx); sum += ex[k]; }
    // deterministic counter-based hash -> uniform
    unsigned s = (unsigned)(t * 0x9E3779B9u) ^ (unsigned)(b * 0x85EBCA6Bu)
               ^ (unsigned)(c * 0xC2B2AE35u) ^ 0x12345u;
    s ^= s >> 16; s *= 0x7FEB352Du; s ^= s >> 15; s *= 0x846CA68Bu; s ^= s >> 16;
    float u = (float)(s >> 8) * (1.0f / 16777216.0f) * sum;
    int idx = 0; float acc = 0.0f;
    for (int k = 0; k < wm::NK_; ++k) { acc += ex[k]; if (u <= acc) { idx = k; break; } idx = k; }
    float lse = logf(sum) + mx;
    size_t pbase = ((size_t)(b * wm::T_ + t)) * wm::LAT_ + c * wm::NK_;
    size_t fbase = ((size_t)(b * wm::T_ + t)) * wm::FULL_ + wm::R_ + c * wm::NK_;
    for (int k = 0; k < wm::NK_; ++k) {
        post_out[pbase + k] = lg[k] - lse;
        float oh = (k == idx) ? 1.0f : 0.0f;   // straight-through forward == one-hot
        fulls_out[fbase + k] = oh;
        xfull[b * wm::FULL_ + wm::R_ + c * wm::NK_ + k] = (__bf16)oh;
    }
}

// ---- copy pre-update h into xfull (bf16) + fulls/recs (f32) ------------
__global__ void k_build_full(const float* __restrict__ h, int t,
                             __bf16* __restrict__ xfull,
                             float* __restrict__ fulls_out,
                             float* __restrict__ recs_out)
{
    int gid = blockIdx.x * blockDim.x + threadIdx.x;
    if (gid >= wm::B_ * wm::R_) return;
    int b = gid / wm::R_, c = gid % wm::R_;
    float hv = h[gid];
    xfull[b * wm::FULL_ + c] = (__bf16)hv;
    fulls_out[((size_t)(b * wm::T_ + t)) * wm::FULL_ + c] = hv;
    recs_out[((size_t)(b * wm::T_ + t)) * wm::R_ + c] = hv;
}

// ---- GRU gates + state update (adds 18-row action correction in VALU) --
__global__ void k_gru_update(const float* __restrict__ gx,
                             const float* __restrict__ gh,
                             const float* __restrict__ gbih,
                             const float* __restrict__ gbhh,
                             const float* __restrict__ gwih_f32,
                             const float* __restrict__ act,
                             const int* __restrict__ dones, int t,
                             float* __restrict__ h)
{
    int tid = blockIdx.x * blockDim.x + threadIdx.x;
    if (tid >= wm::B_ * wm::R_) return;
    int b = tid / wm::R_, r = tid % wm::R_;
    int n = r / wm::BS_, j = r % wm::BS_;
    size_t gb = (size_t)b * wm::NTOTG_ + (size_t)n * wm::G3_;
    size_t bb = (size_t)n * wm::G3_;
    float rx = gx[gb + j]            + gbih[bb + j];
    float zx = gx[gb + wm::BS_ + j]  + gbih[bb + wm::BS_ + j];
    float nx = gx[gb + 2*wm::BS_ + j]+ gbih[bb + 2*wm::BS_ + j];
    float rh = gh[gb + j]            + gbhh[bb + j];
    float zh = gh[gb + wm::BS_ + j]  + gbhh[bb + wm::BS_ + j];
    float nh = gh[gb + 2*wm::BS_ + j]+ gbhh[bb + 2*wm::BS_ + j];
    // action rows (5120..5137) of gwih, f32 path
    const float* abase = act + ((size_t)(b * wm::T_ + t)) * wm::A_;
#pragma unroll
    for (int a = 0; a < wm::A_; ++a) {
        float av = abase[a];
        const float* wr = gwih_f32 + ((size_t)n * wm::GIN_ + wm::FULL_ + a) * wm::G3_;
        rx += av * wr[j];
        zx += av * wr[wm::BS_ + j];
        nx += av * wr[2*wm::BS_ + j];
    }
    float rg = sigf(rx + rh);
    float zg = sigf(zx + zh);
    float nn = tanhf(nx + rg * nh);
    float hp = h[tid];
    float hn = (1.0f - zg) * nn + zg * hp;
    float nd = 1.0f - (float)dones[b * wm::T_ + t];
    h[tid] = hn * nd;
}

// ---- recs f32 -> bf16 for the prior GEMM -------------------------------
__global__ void k_f32_to_bf16(const float* __restrict__ src, __bf16* __restrict__ dst, size_t n)
{
    size_t gid = (size_t)blockIdx.x * blockDim.x + threadIdx.x;
    if (gid < n) dst[gid] = (__bf16)src[gid];
}

// ---- row-group log-softmax over NK=32 classes --------------------------
__global__ void k_log_softmax32(const float* __restrict__ x, float* __restrict__ out, int rows)
{
    int tid = blockIdx.x * blockDim.x + threadIdx.x;
    if (tid >= rows * wm::NC_) return;
    int m = tid / wm::NC_, c = tid % wm::NC_;
    const float* p = x + (size_t)m * wm::LAT_ + c * wm::NK_;
    float mx = p[0];
    for (int k = 1; k < wm::NK_; ++k) mx = fmaxf(mx, p[k]);
    float sum = 0.0f;
    for (int k = 0; k < wm::NK_; ++k) sum += expf(p[k] - mx);
    float lse = logf(sum) + mx;
    float* o = out + (size_t)m * wm::LAT_ + c * wm::NK_;
    for (int k = 0; k < wm::NK_; ++k) o[k] = p[k] - lse;
}

// ---------------------------------------------------------------------------
extern "C" void kernel_launch(void* const* d_in, const int* in_sizes, int n_in,
                              void* d_out, int out_size, void* d_ws, size_t ws_size,
                              hipStream_t stream)
{
    using namespace wm;
    const float* obs   = (const float*)d_in[0];
    const float* act   = (const float*)d_in[1];
    const int*   dones = (const int*)  d_in[2];
    const float* pw1 = (const float*)d_in[3];  const float* pb1 = (const float*)d_in[4];
    const float* pw2 = (const float*)d_in[5];  const float* pb2 = (const float*)d_in[6];
    const float* qw1 = (const float*)d_in[7];  const float* qb1 = (const float*)d_in[8];
    const float* qw2 = (const float*)d_in[9];  const float* qb2 = (const float*)d_in[10];
    const float* gwih = (const float*)d_in[11];
    const float* gwhh = (const float*)d_in[12];
    const float* gbih = (const float*)d_in[13];
    const float* gbhh = (const float*)d_in[14];

    float* outF     = (float*)d_out;
    float* fullsOut = outF + OFF_FULLS;
    float* recsOut  = outF + OFF_RECS;
    float* postOut  = outF + OFF_POST;
    float* priorOut = outF + OFF_PRIOR;

    // ---- workspace carve-up (~170 MB needed; weights ~161MB -> L2-resident)
    char* base = (char*)d_ws;
    size_t off = 0;
    auto carve = [&](size_t bytes) { void* p = base + off; off = (off + bytes + 255) & ~(size_t)255; return p; };
    __bf16* wPw1  = (__bf16*)carve((size_t)FULL_ * 1024 * 2);
    __bf16* wPw2  = (__bf16*)carve((size_t)1024 * 1024 * 2);
    __bf16* wQw1  = (__bf16*)carve((size_t)R_ * 1024 * 2);
    __bf16* wQw2  = (__bf16*)carve((size_t)1024 * 1024 * 2);
    __bf16* wGwih = (__bf16*)carve((size_t)FULL_ * NTOTG_ * 2);
    __bf16* wGwhh = (__bf16*)carve((size_t)BS_ * NTOTG_ * 2);
    float*  hBuf  = (float*) carve((size_t)B_ * R_ * 4);
    __bf16* xpost = (__bf16*)carve((size_t)B_ * FULL_ * 2);
    __bf16* y1bf  = (__bf16*)carve((size_t)B_ * 1024 * 2);
    float*  logit = (float*) carve((size_t)B_ * LAT_ * 4);
    __bf16* xfull = (__bf16*)carve((size_t)B_ * FULL_ * 2);
    float*  gxB   = (float*) carve((size_t)B_ * NTOTG_ * 4);
    float*  ghB   = (float*) carve((size_t)B_ * NTOTG_ * 4);
    __bf16* recsB = (__bf16*)carve((size_t)B_ * T_ * R_ * 2);
    __bf16* phbf  = (__bf16*)carve((size_t)B_ * T_ * 1024 * 2);
    float*  plog  = (float*) carve((size_t)B_ * T_ * LAT_ * 4);
    (void)ws_size; (void)in_sizes; (void)n_in; (void)out_size;

    auto gridFor = [](size_t n) { return dim3((unsigned)((n + 255) / 256)); };

    // ---- pack all weights to bf16 fragment tiles (re-done every call;
    //      deterministic, and they stay hot in the 192MB L2 afterwards)
    k_pack_b<<<gridFor((size_t)FULL_ * 1024), 256, 0, stream>>>(pw1, wPw1, FULL_, 1024, 1024, 0, 1024);
    k_pack_b<<<gridFor((size_t)1024 * 1024), 256, 0, stream>>>(pw2, wPw2, 1024, 1024, 1024, 0, 1024);
    k_pack_b<<<gridFor((size_t)R_ * 1024),   256, 0, stream>>>(qw1, wQw1, R_, 1024, 1024, 0, 1024);
    k_pack_b<<<gridFor((size_t)1024 * 1024), 256, 0, stream>>>(qw2, wQw2, 1024, 1024, 1024, 0, 1024);
    k_pack_b<<<gridFor((size_t)FULL_ * NTOTG_), 256, 0, stream>>>(gwih, wGwih, FULL_, NTOTG_, G3_, (long long)GIN_ * G3_, G3_);
    k_pack_b<<<gridFor((size_t)BS_ * NTOTG_),   256, 0, stream>>>(gwhh, wGwhh, BS_, NTOTG_, G3_, (long long)BS_ * G3_, G3_);

    hipMemsetAsync(hBuf, 0, (size_t)B_ * R_ * 4, stream);   // h0 = 0

    // ---- the sequential scan ------------------------------------------
    for (int t = 0; t < T_; ++t) {
        // posterior MLP
        k_build_xpost<<<gridFor((size_t)B_ * FULL_), 256, 0, stream>>>(hBuf, obs, t, xpost);
        k_gemm_wmma<true ><<<dim3(1024 / 16 / 8, 1), 256, 0, stream>>>(
            xpost, FULL_, wPw1, pb1, y1bf, FULL_, 1024, 1024, 0);
        k_gemm_wmma<false><<<dim3(1024 / 16 / 8, 1), 256, 0, stream>>>(
            y1bf, 1024, wPw2, pb2, logit, 1024, LAT_, LAT_, 0);
        // sample latent + post log-softmax; copy h into full/recs
        k_sample<<<gridFor((size_t)B_ * NC_), 256, 0, stream>>>(logit, t, postOut, fullsOut, xfull);
        k_build_full<<<gridFor((size_t)B_ * R_), 256, 0, stream>>>(hBuf, t, xfull, fullsOut, recsOut);
        // GRU: input GEMM [16,5120]x[5120,12288], recurrent [16,512]x[512,1536] per block
        k_gemm_wmma<false><<<dim3(NTOTG_ / 16 / 8, 1), 256, 0, stream>>>(
            xfull, FULL_, wGwih, nullptr, gxB, FULL_, NTOTG_, NTOTG_, 0);
        k_gemm_wmma<false><<<dim3(NTOTG_ / 16 / 8, 1), 256, 0, stream>>>(
            xfull, FULL_, wGwhh, nullptr, ghB, BS_, NTOTG_, G3_, BS_);
        k_gru_update<<<gridFor((size_t)B_ * R_), 256, 0, stream>>>(
            gxB, ghB, gbih, gbhh, gwih, act, dones, t, hBuf);
    }

    // ---- prior MLP over all (b,t) rows --------------------------------
    k_f32_to_bf16<<<gridFor((size_t)B_ * T_ * R_), 256, 0, stream>>>(recsOut, recsB, (size_t)B_ * T_ * R_);
    k_gemm_wmma<true ><<<dim3(1024 / 16 / 8, (B_ * T_) / 16), 256, 0, stream>>>(
        recsB, R_, wQw1, qb1, phbf, R_, 1024, 1024, 0);
    k_gemm_wmma<false><<<dim3(1024 / 16 / 8, (B_ * T_) / 16), 256, 0, stream>>>(
        phbf, 1024, wQw2, qb2, plog, 1024, LAT_, LAT_, 0);
    k_log_softmax32<<<gridFor((size_t)B_ * T_ * NC_), 256, 0, stream>>>(plog, priorOut, B_ * T_);
}
